// SVDLinear_33303176413292
// MI455X (gfx1250) — matrix operand order repaired and checked by reading, not compiled, not run on hardware
//
#include <hip/hip_runtime.h>
#include <hip/hip_bf16.h>

// ---- problem constants (from reference) ----
#define IN_F   4096
#define OUT_F  4096
#define RANK   1024
#define MTOT   8192   // BATCH(4) * SEQ(2048)

// ---- tile constants ----
#define BM 128        // block M tile (4 waves x 32 rows)
#define BN 64         // block N tile
#define BK 32         // K step (bf16 WMMA K)
#define LDSS 40       // bf16 elems per LDS row (32 + 8 pad) -> 80B rows, 16B-aligned

typedef __attribute__((ext_vector_type(16))) __bf16   v16bf;
typedef __attribute__((ext_vector_type(8)))  __bf16   v8bf;
typedef __attribute__((ext_vector_type(8)))  float    v8f;
typedef __attribute__((ext_vector_type(4)))  unsigned v4u;
typedef __attribute__((ext_vector_type(2)))  unsigned v2u;

__device__ __forceinline__ __bf16 f2bf(float f) { return (__bf16)f; }

// pack two floats into one dword of bf16 pair (lo, hi) -> v_cvt_pk_bf16_f32
__device__ __forceinline__ unsigned pk(float a, float b) {
  union { __bf16 h; unsigned short u; } x, y;
  x.h = (__bf16)a; y.h = (__bf16)b;
  return (unsigned)x.u | ((unsigned)y.u << 16);
}

__device__ __forceinline__ float fidx(const float4& v, int j) {
  return j == 0 ? v.x : j == 1 ? v.y : j == 2 ? v.z : v.w;
}

// Workgroup barrier that publishes only LDS (no global/loadcnt drain):
// release/acquire fences restricted to the "local" address space keep the
// in-flight next-tile global loads outstanding across the barrier.
__device__ __forceinline__ void lds_barrier() {
  __builtin_amdgcn_fence(__ATOMIC_RELEASE, "workgroup", "local");
  __builtin_amdgcn_s_barrier_signal(-1);
  __builtin_amdgcn_s_barrier_wait(-1);
  __builtin_amdgcn_fence(__ATOMIC_ACQUIRE, "workgroup", "local");
}

// Async HBM -> LDS copy, 16 bytes per lane, tracked by ASYNCcnt.
// LDS byte address = low 32 bits of the generic shared pointer (LDS aperture
// keeps the offset in addr[31:0], ISA 10.2).
__device__ __forceinline__ void async_copy16(const void* g, void* lds_generic) {
  unsigned l = (unsigned)(unsigned long long)lds_generic;
  asm volatile("global_load_async_to_lds_b128 %0, %1, off"
               :: "v"(l), "v"(g) : "memory");
}
__device__ __forceinline__ void wait_async0() {
  asm volatile("s_wait_asynccnt 0x0" ::: "memory");
}

// A fragment (16x32 bf16): lane l holds row m=l%16.
// Lanes 0-15: halfwords 0-7 -> K=0..7, 8-15 -> K=16..23
// Lanes 16-31: halfwords 0-7 -> K=8..15, 8-15 -> K=24..31
__device__ __forceinline__ v16bf load_a_frag(const __bf16* lds, int lane, int row) {
  const int b0 = (lane < 16) ? 0 : 8;
  v8bf lo = *reinterpret_cast<const v8bf*>(lds + row * LDSS + b0);
  v8bf hi = *reinterpret_cast<const v8bf*>(lds + row * LDSS + b0 + 16);
  v16bf a;
#pragma unroll
  for (int i = 0; i < 8; ++i) { a[i] = lo[i]; a[i + 8] = hi[i]; }
  return a;
}

// B fragment (32x16 bf16): lane l holds column n=l%16; lanes 0-15 K=0..7,16..23,
// lanes 16-31 K=8..15,24..31. LDS B tile stored transposed: ldsB[n][k].
__device__ __forceinline__ v16bf load_b_frag(const __bf16* lds, int lane, int col) {
  const int kb = (lane < 16) ? 0 : 8;
  v8bf lo = *reinterpret_cast<const v8bf*>(lds + col * LDSS + kb);
  v8bf hi = *reinterpret_cast<const v8bf*>(lds + col * LDSS + kb + 16);
  v16bf b;
#pragma unroll
  for (int i = 0; i < 8; ++i) { b[i] = lo[i]; b[i + 8] = hi[i]; }
  return b;
}

// ============================================================================
// Kernel 1: H[m, r] = (sum_k X[m,k] * Vt[k,r]) * S[r]   (bf16 out to ws)
// Double-buffered LDS + ping-pong register sets; LDS-only barrier keeps the
// next-tile global loads in flight across the WMMA block.
// grid = (RANK/BN, MTOT/BM), block = 128 (4 waves, 32x64 per wave)
// ============================================================================
__global__ __launch_bounds__(128)
void svd_gemm1(const float* __restrict__ X, const float* __restrict__ Vt,
               const float* __restrict__ S, __bf16* __restrict__ H) {
  __shared__ __bf16 ldsA[2][BM * LDSS];   // 2 x 10240 B
  __shared__ __bf16 ldsB[2][BN * LDSS];   // 2 x  5120 B

  const int tid  = threadIdx.x;
  const int wave = tid >> 5;
  const int lane = tid & 31;
  const int n0 = blockIdx.x * BN;
  const int m0 = blockIdx.y * BM;

  v8f acc[8] = {};   // [at*4 + nt]

  const int arow = tid >> 1;            // 0..63 (+64 on pass 1)
  const int acol = (tid & 1) * 16;      // 0 or 16
  const int bk4 = (tid >> 4) * 4;       // 0,4,...,28
  const int bn4 = (tid & 15) * 4;       // 0,4,...,60

  float4 raA[8], rbA[4];   // register set A
  float4 raB[8], rbB[4];   // register set B

  auto load_regs = [&](int kk, float4 (&ra)[8], float4 (&rb)[4]) {
#pragma unroll
    for (int pass = 0; pass < 2; ++pass) {
      const float4* xa = reinterpret_cast<const float4*>(
          X + (size_t)(m0 + arow + pass * 64) * IN_F + kk + acol);
#pragma unroll
      for (int j = 0; j < 4; ++j) ra[pass * 4 + j] = xa[j];
    }
#pragma unroll
    for (int i = 0; i < 4; ++i)
      rb[i] = *reinterpret_cast<const float4*>(
          Vt + (size_t)(kk + bk4 + i) * RANK + n0 + bn4);
  };

  auto store_lds = [&](int buf, float4 (&ra)[8], float4 (&rb)[4]) {
#pragma unroll
    for (int pass = 0; pass < 2; ++pass) {
      float4 v0 = ra[pass * 4 + 0], v1 = ra[pass * 4 + 1];
      float4 v2 = ra[pass * 4 + 2], v3 = ra[pass * 4 + 3];
      v4u p0 = { pk(v0.x, v0.y), pk(v0.z, v0.w), pk(v1.x, v1.y), pk(v1.z, v1.w) };
      v4u p1 = { pk(v2.x, v2.y), pk(v2.z, v2.w), pk(v3.x, v3.y), pk(v3.z, v3.w) };
      __bf16* dst = ldsA[buf] + (arow + pass * 64) * LDSS + acol;
      *reinterpret_cast<v4u*>(dst)     = p0;
      *reinterpret_cast<v4u*>(dst + 8) = p1;
    }
#pragma unroll
    for (int j = 0; j < 4; ++j) {
      v2u p = { pk(fidx(rb[0], j), fidx(rb[1], j)),
                pk(fidx(rb[2], j), fidx(rb[3], j)) };
      *reinterpret_cast<v2u*>(ldsB[buf] + (bn4 + j) * LDSS + bk4) = p;
    }
  };

  auto compute = [&](int buf) {
    const __bf16* A = ldsA[buf];
    const __bf16* B = ldsB[buf];
    v16bf a0 = load_a_frag(A, lane, wave * 32 + (lane & 15));
    v16bf a1 = load_a_frag(A, lane, wave * 32 + 16 + (lane & 15));
#pragma unroll
    for (int nt = 0; nt < 4; ++nt) {
      v16bf b = load_b_frag(B, lane, nt * 16 + (lane & 15));
      acc[nt]     = __builtin_amdgcn_wmma_f32_16x16x32_bf16(
          false, a0, false, b, (short)0, acc[nt],     false, false);
      acc[4 + nt] = __builtin_amdgcn_wmma_f32_16x16x32_bf16(
          false, a1, false, b, (short)0, acc[4 + nt], false, false);
    }
  };

  const int NIT = IN_F / BK;   // 128 (even)
  load_regs(0, raA, rbA);
  for (int it = 0; it < NIT; it += 2) {
    // body 0: consume set A into buf0, fetch next into set B
    store_lds(0, raA, rbA);
    if (it + 1 < NIT) load_regs((it + 1) * BK, raB, rbB);
    lds_barrier();
    compute(0);
    // body 1: consume set B into buf1, fetch next into set A
    store_lds(1, raB, rbB);
    if (it + 2 < NIT) load_regs((it + 2) * BK, raA, rbA);
    lds_barrier();
    compute(1);
  }

  // epilogue: scale by S[r], store bf16. C/D: elem v -> m=v+8*(lane>=16), n=lane%16
#pragma unroll
  for (int at = 0; at < 2; ++at) {
#pragma unroll
    for (int nt = 0; nt < 4; ++nt) {
      const int n = n0 + nt * 16 + (lane & 15);
      const float s = S[n];
#pragma unroll
      for (int v = 0; v < 8; ++v) {
        const int m = m0 + wave * 32 + at * 16 + v + ((lane < 16) ? 0 : 8);
        H[(size_t)m * RANK + n] = f2bf(acc[at * 4 + nt][v] * s);
      }
    }
  }
}

// ============================================================================
// Kernel 2: Y[m, o] = sum_r H[m,r] * U[o,r] + bias[o]
// A tile (bf16 pass-through) via global_load_async_to_lds_b128 (ASYNCcnt);
// B tile (U rows, contiguous) ping-pong register pipelined with bf16 packing.
// grid = (OUT_F/BN, MTOT/BM), block = 128 (4 waves, 32x64 per wave)
// ============================================================================
__global__ __launch_bounds__(128)
void svd_gemm2(const __bf16* __restrict__ H, const float* __restrict__ U,
               const float* __restrict__ bias, float* __restrict__ Y) {
  __shared__ __bf16 ldsA[2][BM * LDSS];
  __shared__ __bf16 ldsB[2][BN * LDSS];

  const int tid  = threadIdx.x;
  const int wave = tid >> 5;
  const int lane = tid & 31;
  const int n0 = blockIdx.x * BN;   // output feature
  const int m0 = blockIdx.y * BM;

  v8f acc[8] = {};

  const int arow = tid >> 1;          // 0..63 (+64 on pass 1)
  const int acol = (tid & 1) * 16;    // 0 or 16
  const int brow = tid >> 1;          // U row = output feature
  const int bcol = (tid & 1) * 16;

  float4 rbA[4], rbB[4];

  auto issue_asyncA = [&](int buf, int kk) {
#pragma unroll
    for (int pass = 0; pass < 2; ++pass) {
      const int r = arow + pass * 64;
      const __bf16* g = H + (size_t)(m0 + r) * RANK + kk + acol;
      __bf16* l = ldsA[buf] + r * LDSS + acol;
      async_copy16(g,     l);
      async_copy16(g + 8, l + 8);
    }
  };
  auto load_regsB = [&](int kk, float4 (&rb)[4]) {
    const float4* ub = reinterpret_cast<const float4*>(
        U + (size_t)(n0 + brow) * RANK + kk + bcol);
#pragma unroll
    for (int i = 0; i < 4; ++i) rb[i] = ub[i];
  };
  auto store_ldsB = [&](int buf, float4 (&rb)[4]) {
    v4u p0 = { pk(rb[0].x, rb[0].y), pk(rb[0].z, rb[0].w),
               pk(rb[1].x, rb[1].y), pk(rb[1].z, rb[1].w) };
    v4u p1 = { pk(rb[2].x, rb[2].y), pk(rb[2].z, rb[2].w),
               pk(rb[3].x, rb[3].y), pk(rb[3].z, rb[3].w) };
    __bf16* dst = ldsB[buf] + brow * LDSS + bcol;
    *reinterpret_cast<v4u*>(dst)     = p0;
    *reinterpret_cast<v4u*>(dst + 8) = p1;
  };
  auto compute = [&](int buf) {
    const __bf16* A = ldsA[buf];
    const __bf16* B = ldsB[buf];
    v16bf a0 = load_a_frag(A, lane, wave * 32 + (lane & 15));
    v16bf a1 = load_a_frag(A, lane, wave * 32 + 16 + (lane & 15));
#pragma unroll
    for (int nt = 0; nt < 4; ++nt) {
      v16bf b = load_b_frag(B, lane, nt * 16 + (lane & 15));
      acc[nt]     = __builtin_amdgcn_wmma_f32_16x16x32_bf16(
          false, a0, false, b, (short)0, acc[nt],     false, false);
      acc[4 + nt] = __builtin_amdgcn_wmma_f32_16x16x32_bf16(
          false, a1, false, b, (short)0, acc[4 + nt], false, false);
    }
  };

  const int NIT = RANK / BK;   // 32 (even)
  issue_asyncA(0, 0);          // prologue: async-stage A(buf0)
  load_regsB(0, rbA);

  for (int it = 0; it < NIT; it += 2) {
    // body 0: buf0 current
    store_ldsB(0, rbA);
    if (it + 1 < NIT) load_regsB((it + 1) * BK, rbB);
    wait_async0();                         // A(buf0) asyncs complete
    lds_barrier();                         // publish A(buf0) + B(buf0)
    if (it + 1 < NIT) issue_asyncA(1, (it + 1) * BK);  // overlap with compute
    compute(0);
    // body 1: buf1 current
    store_ldsB(1, rbB);
    if (it + 2 < NIT) load_regsB((it + 2) * BK, rbA);
    wait_async0();                         // A(buf1) asyncs complete
    lds_barrier();
    if (it + 2 < NIT) issue_asyncA(0, (it + 2) * BK);
    compute(1);
  }

  // epilogue: add bias, store f32
#pragma unroll
  for (int at = 0; at < 2; ++at) {
#pragma unroll
    for (int nt = 0; nt < 4; ++nt) {
      const int n = n0 + nt * 16 + (lane & 15);
      const float bz = bias[n];
#pragma unroll
      for (int v = 0; v < 8; ++v) {
        const int m = m0 + wave * 32 + at * 16 + v + ((lane < 16) ? 0 : 8);
        Y[(size_t)m * OUT_F + n] = acc[at * 4 + nt][v] + bz;
      }
    }
  }
}

extern "C" void kernel_launch(void* const* d_in, const int* in_sizes, int n_in,
                              void* d_out, int out_size, void* d_ws, size_t ws_size,
                              hipStream_t stream) {
  // setup_inputs order: x, U, S, Vt, bias
  const float* x    = (const float*)d_in[0];
  const float* U    = (const float*)d_in[1];
  const float* S    = (const float*)d_in[2];
  const float* Vt   = (const float*)d_in[3];
  const float* bias = (const float*)d_in[4];
  float* out = (float*)d_out;

  // Intermediate H = (x @ Vt) * S as bf16: 8192 * 1024 * 2B = 16 MB in d_ws
  __bf16* H = (__bf16*)d_ws;

  dim3 blk(128, 1, 1);
  dim3 g1(RANK / BN, MTOT / BM, 1);   // 16 x 64
  svd_gemm1<<<g1, blk, 0, stream>>>(x, Vt, S, H);

  dim3 g2(OUT_F / BN, MTOT / BM, 1);  // 64 x 64
  svd_gemm2<<<g2, blk, 0, stream>>>(H, U, bias, out);
}
